// UberCRSN_35656818492159
// MI455X (gfx1250) — matrix operand order repaired and checked by reading, not compiled, not run on hardware
//
#include <hip/hip_runtime.h>

#define B_  2048
#define D_  1024
#define D2_ 2048
#define S_  8192
#define V_  50257
#define TMX 8

typedef __attribute__((ext_vector_type(16))) _Float16 v16h;
typedef __attribute__((ext_vector_type(8)))  _Float16 v8h;
typedef __attribute__((ext_vector_type(4)))  _Float16 v4h;
typedef __attribute__((ext_vector_type(8)))  float    v8f;

// ---------------------------------------------------------------------------
// Generic GEMM: C[M,N] = A[M,K] * B[N,K]^T (+bias), f32 in/out, f16 WMMA math.
// Block tile 128x128, K-stage 64 (= 2 WMMA K-steps). 256 threads = 8 waves in
// 4(M) x 2(N); each wave computes a 32x64 tile via 8 accumulators, issuing
// 16x v_wmma_f32_16x16x32_f16 per LDS staging round (4x density vs 32-K stage).
// GUARD_N=false instantiation has straight-line, exec-full staging (all in-loop
// GEMMs are tile-divisible); GUARD_N=true only for the decoder (N=50257).
// ---------------------------------------------------------------------------
#define BM 128
#define BN 128
#define BK 64
#define LDS_STRIDE 72   // halves; 144B rows keep 16B alignment for v8h loads

__device__ __forceinline__ v16h ucrsn_combine(v8h lo, v8h hi) {
  v16h r;
#pragma unroll
  for (int i = 0; i < 8; ++i) { r[i] = lo[i]; r[i + 8] = hi[i]; }
  return r;
}

template <bool GUARD_N>
__global__ __launch_bounds__(256) void ucrsn_gemm_f16(
    const float* __restrict__ A, const float* __restrict__ Bm,
    float* __restrict__ C, const float* __restrict__ bias,
    int M, int N, int K)
{
  __shared__ _Float16 shA[BM * LDS_STRIDE];
  __shared__ _Float16 shB[BN * LDS_STRIDE];

  const int tid    = threadIdx.x;
  const int lane   = tid & 31;
  const int wave   = tid >> 5;
  const int wm     = wave & 3;    // 4 waves down M (32 rows each)
  const int wn     = wave >> 2;   // 2 waves across N (64 cols each)
  const int blockM = blockIdx.y * BM;
  const int blockN = blockIdx.x * BN;

  v8f acc[8];
#pragma unroll
  for (int i = 0; i < 8; ++i)
#pragma unroll
    for (int e = 0; e < 8; ++e) acc[i][e] = 0.0f;

  const int laneRow = lane & 15;
  const int laneHi  = lane >> 4;

  // Staging decomposition: 128x64 f32 tile = 2048 float4; 8 float4 per thread.
  const int stRow = tid >> 4;        // 0..15 base row (steps of 16 rows/iter? no:)
  const int stK4  = tid & 15;        // float4 column 0..15 (64 floats/row)

  for (int kt = 0; kt < K; kt += BK) {
    if (kt + BK < K) {
      __builtin_prefetch(A + (size_t)(blockM + (tid & 127)) * K + kt + BK, 0, 1);
      int pn = blockN + (tid & 127);
      if (!GUARD_N || pn < N)
        __builtin_prefetch(Bm + (size_t)pn * K + kt + BK, 0, 1);
    }
    __syncthreads();
    // Stage A tile: 128x64 f32 -> f16 (8 float4 per thread, rows stRow+16*i).
#pragma unroll
    for (int i = 0; i < 8; ++i) {
      int row = stRow + i * 16;
      float4 v = *(const float4*)(A + (size_t)(blockM + row) * K + kt + stK4 * 4);
      v4h h; h[0] = (_Float16)v.x; h[1] = (_Float16)v.y;
             h[2] = (_Float16)v.z; h[3] = (_Float16)v.w;
      *(v4h*)&shA[row * LDS_STRIDE + stK4 * 4] = h;
    }
    // Stage B tile: 128x64 f32 -> f16.
#pragma unroll
    for (int i = 0; i < 8; ++i) {
      int row = stRow + i * 16;
      int gn  = blockN + row;
      float4 v = {0.f, 0.f, 0.f, 0.f};
      if (!GUARD_N || gn < N)
        v = *(const float4*)(Bm + (size_t)gn * K + kt + stK4 * 4);
      v4h h; h[0] = (_Float16)v.x; h[1] = (_Float16)v.y;
             h[2] = (_Float16)v.z; h[3] = (_Float16)v.w;
      *(v4h*)&shB[row * LDS_STRIDE + stK4 * 4] = h;
    }
    __syncthreads();

#pragma unroll
    for (int ks = 0; ks < 2; ++ks) {   // two 32-wide WMMA K-steps per stage
      int kofs = ks * 32;
      // A fragments (ISA 7.12.2): lanes 0-15 rows M=laneRow hold K {kb..kb+7,
      // 16+kb..16+kb+7}; lanes 16-31 same rows, kb=8.
      int ar0 = wm * 32 + laneRow;
      int akb = kofs + laneHi * 8;
      v16h a0 = ucrsn_combine(*(const v8h*)&shA[ar0 * LDS_STRIDE + akb],
                              *(const v8h*)&shA[ar0 * LDS_STRIDE + 16 + akb]);
      v16h a1 = ucrsn_combine(*(const v8h*)&shA[(ar0 + 16) * LDS_STRIDE + akb],
                              *(const v8h*)&shA[(ar0 + 16) * LDS_STRIDE + 16 + akb]);
      // B fragments: lanes 0-15 N=laneRow hold K {0..7,16..23}; lanes 16-31 K {8..15,24..31}.
      int bk0 = kofs + laneHi * 8;
      v16h b[4];
#pragma unroll
      for (int tj = 0; tj < 4; ++tj) {
        int bn = wn * 64 + tj * 16 + laneRow;
        b[tj] = ucrsn_combine(*(const v8h*)&shB[bn * LDS_STRIDE + bk0],
                              *(const v8h*)&shB[bn * LDS_STRIDE + bk0 + 16]);
      }
#pragma unroll
      for (int tj = 0; tj < 4; ++tj) {
        acc[tj]     = __builtin_amdgcn_wmma_f32_16x16x32_f16(
            false, a0, false, b[tj], (short)0, acc[tj], false, false);
        acc[4 + tj] = __builtin_amdgcn_wmma_f32_16x16x32_f16(
            false, a1, false, b[tj], (short)0, acc[4 + tj], false, false);
      }
    }
  }

  // Epilogue: C/D layout — lane<16: N=laneRow, rows M=v; lane>=16: rows M=v+8.
  const int rowOff = (lane < 16) ? 0 : 8;
#pragma unroll
  for (int ti = 0; ti < 2; ++ti)
#pragma unroll
    for (int tj = 0; tj < 4; ++tj) {
      int tm  = blockM + wm * 32 + ti * 16;
      int tn  = blockN + wn * 64 + tj * 16;
      int col = tn + laneRow;
      float bv = 0.0f;
      if (bias != nullptr && (!GUARD_N || col < N)) bv = bias[col];
      v8f a = acc[ti * 4 + tj];
#pragma unroll
      for (int r = 0; r < 8; ++r) {
        int m = tm + r + rowOff;
        if (!GUARD_N || col < N) C[(size_t)m * N + col] = a[r] + bv;
      }
    }
}

// ---------------------------------------------------------------------------
// Init: complex embedding, zero weighted accumulators and ACT state.
// ---------------------------------------------------------------------------
__global__ void ucrsn_init(const int* __restrict__ ids, const int* __restrict__ pidx,
                           const int* __restrict__ aidx,
                           const float* __restrict__ emb_mag,
                           const float* __restrict__ emb_phase,
                           const float* __restrict__ persp_emb,
                           const float* __restrict__ audio_emb,
                           float* __restrict__ z, float* __restrict__ zw,
                           float* __restrict__ halting, float* __restrict__ remain,
                           float* __restrict__ accs)
{
  int gid = blockIdx.x * blockDim.x + threadIdx.x;
  if (gid < B_) { halting[gid] = 0.f; remain[gid] = 1.f; }
  if (gid == 0) { accs[0] = 0.f; accs[1] = 0.f; }
  if (gid >= B_ * D_) return;
  int b = gid >> 10, j = gid & 1023;
  int id = ids[b];
  float r  = emb_mag[(size_t)id * D_ + j];
  float th = emb_phase[(size_t)id * D_ + j];
  float zr = r * cosf(th) + persp_emb[(size_t)pidx[b] * D_ + j]
                          + audio_emb[(size_t)aidx[b] * D_ + j];
  float zi = r * sinf(th);
  size_t base = (size_t)b * D2_;
  z[base + j] = zr;  z[base + D_ + j] = zi;
  zw[base + j] = 0.f; zw[base + D_ + j] = 0.f;
}

// Build stacked rotation matrix: rows 0..D-1 = [Wr | -Wi]; rows D..2D-1 = [Wi | Wr].
__global__ void ucrsn_bstack(const float* __restrict__ Wr, const float* __restrict__ Wi,
                             float* __restrict__ Bs)
{
  int gid = blockIdx.x * blockDim.x + threadIdx.x;
  if (gid >= D_ * D_) return;
  int n = gid >> 10, k = gid & 1023;
  float wr = Wr[gid], wi = Wi[gid];
  Bs[(size_t)n * D2_ + k]             =  wr;
  Bs[(size_t)n * D2_ + D_ + k]        = -wi;
  Bs[(size_t)(D_ + n) * D2_ + k]      =  wi;
  Bs[(size_t)(D_ + n) * D2_ + D_ + k] =  wr;
}

__global__ __launch_bounds__(256) void ucrsn_cbsq(const float* __restrict__ cb,
                                                  float* __restrict__ cb_sq)
{
  __shared__ float red[256];
  int s = blockIdx.x;
  float acc = 0.f;
  for (int j = threadIdx.x; j < D2_; j += 256) {
    float v = cb[(size_t)s * D2_ + j];
    acc += v * v;
  }
  red[threadIdx.x] = acc; __syncthreads();
  for (int o = 128; o > 0; o >>= 1) {
    if (threadIdx.x < o) red[threadIdx.x] += red[threadIdx.x + o];
    __syncthreads();
  }
  if (threadIdx.x == 0) cb_sq[s] = red[0];
}

// ---------------------------------------------------------------------------
// Per-row recurrent nonlinearity: magnitude LayerNorm + modReLU + halt prob +
// sensory-biased z_flat. One workgroup per batch row.
// ---------------------------------------------------------------------------
__global__ __launch_bounds__(256) void ucrsn_recurrent(
    const float* __restrict__ l, const float* __restrict__ ln_scale,
    const float* __restrict__ ln_shift, const float* __restrict__ mod_bias,
    const float* __restrict__ halt_W, const float* __restrict__ halt_b,
    const int* __restrict__ olf_idx, const int* __restrict__ gust_idx,
    const float* __restrict__ olf_emb, const float* __restrict__ gust_emb,
    float* __restrict__ nr_buf, float* __restrict__ ni_buf,
    float* __restrict__ z_flat, float* __restrict__ p_halt, int t)
{
  __shared__ float red[256];
  __shared__ float sh_mean, sh_rstd;
  int b = blockIdx.x, tid = threadIdx.x;
  size_t base = (size_t)b * D2_;

  float s1 = 0.f, s2 = 0.f;
  for (int j = tid; j < D_; j += 256) {
    float lr = l[base + j], li = l[base + D_ + j];
    float mag = sqrtf(lr * lr + li * li) + 1e-6f;
    s1 += mag; s2 += mag * mag;
  }
  red[tid] = s1; __syncthreads();
  for (int o = 128; o > 0; o >>= 1) { if (tid < o) red[tid] += red[tid + o]; __syncthreads(); }
  float s1t = red[0]; __syncthreads();
  red[tid] = s2; __syncthreads();
  for (int o = 128; o > 0; o >>= 1) { if (tid < o) red[tid] += red[tid + o]; __syncthreads(); }
  float s2t = red[0]; __syncthreads();
  if (tid == 0) {
    float mean = s1t / (float)D_;
    float var  = (s2t - s1t * s1t / (float)D_) / (float)(D_ - 1);
    sh_mean = mean; sh_rstd = rsqrtf(var + 1e-6f);
  }
  __syncthreads();
  float mean = sh_mean, rstd = sh_rstd;

  int oi = olf_idx[t * B_ + b], gi = gust_idx[t * B_ + b];
  float hacc = 0.f;
  for (int j = tid; j < D_; j += 256) {
    float lr = l[base + j], li = l[base + D_ + j];
    float mag0 = sqrtf(lr * lr + li * li);
    float mag  = mag0 + 1e-6f;
    float norm = (mag - mean) * rstd * ln_scale[j] + ln_shift[j];
    float inv  = 1.0f / fmaxf(mag0, 1e-30f);
    float nr0 = norm * lr * inv, ni0 = norm * li * inv;
    float m = sqrtf(nr0 * nr0 + ni0 * ni0) + 1e-6f;
    float sfac = fmaxf(m + mod_bias[j], 0.f) / m;
    float nr = nr0 * sfac, ni = ni0 * sfac;
    nr_buf[(size_t)b * D_ + j] = nr;
    ni_buf[(size_t)b * D_ + j] = ni;
    hacc += nr * halt_W[j] + ni * halt_W[D_ + j];
    float sr = olf_emb[(size_t)oi * D2_ + j]      + gust_emb[(size_t)gi * D2_ + j];
    float si = olf_emb[(size_t)oi * D2_ + D_ + j] + gust_emb[(size_t)gi * D2_ + D_ + j];
    z_flat[base + j]      = nr + 0.1f * (float)(t + 1) + sr;
    z_flat[base + D_ + j] = ni + si;
  }
  red[tid] = hacc; __syncthreads();
  for (int o = 128; o > 0; o >>= 1) { if (tid < o) red[tid] += red[tid + o]; __syncthreads(); }
  if (tid == 0) p_halt[b] = 1.0f / (1.0f + expf(-(red[0] + halt_b[0])));
}

// ---------------------------------------------------------------------------
// Fused distance argmin (|z|^2 term dropped: constant per row) + adjacency
// bias + ACT halting update + ponder accumulation. One workgroup per row.
// ---------------------------------------------------------------------------
__global__ __launch_bounds__(256) void ucrsn_argmin_act(
    const float* __restrict__ G, const float* __restrict__ cb_sq,
    const float* __restrict__ adjacency, int* __restrict__ idxbuf,
    const float* __restrict__ p_halt, float* __restrict__ p_buf,
    float* __restrict__ halting, float* __restrict__ remain,
    float* __restrict__ accs, int t)
{
  __shared__ float rd[256];
  __shared__ int   ri[256];
  int b = blockIdx.x, tid = threadIdx.x;
  int prev = idxbuf[b];  // only dereferenced when t > 0
  float best = 3.4e38f; int besti = 0x7fffffff;
  for (int s = tid; s < S_; s += 256) {
    float dv = cb_sq[s] - 2.0f * G[(size_t)b * S_ + s];
    if (t > 0) {
      float sig = 1.0f / (1.0f + expf(-adjacency[(size_t)prev * S_ + s]));
      dv -= 0.8f * fminf(sig, 0.3f);
    }
    if (dv < best || (dv == best && s < besti)) { best = dv; besti = s; }
  }
  rd[tid] = best; ri[tid] = besti; __syncthreads();
  for (int o = 128; o > 0; o >>= 1) {
    if (tid < o) {
      float ov = rd[tid + o]; int oj = ri[tid + o];
      if (ov < rd[tid] || (ov == rd[tid] && oj < ri[tid])) { rd[tid] = ov; ri[tid] = oj; }
    }
    __syncthreads();
  }
  if (tid == 0) {
    idxbuf[b] = ri[0];
    float h = halting[b];
    float still = (h < 0.999f) ? 1.0f : 0.0f;
    float p = (t == TMX - 1) ? remain[b] : p_halt[b] * still;
    p_buf[b]   = p;
    halting[b] = h + p;
    remain[b]  = remain[b] - p;
    atomicAdd(&accs[0], still * (1.0f / (float)B_));
  }
}

// ---------------------------------------------------------------------------
// VQ straight-through state update, ACT-weighted output accumulation, VQ loss.
// ---------------------------------------------------------------------------
__global__ __launch_bounds__(256) void ucrsn_update(
    const float* __restrict__ codebook, const int* __restrict__ idxbuf,
    const float* __restrict__ nr_buf, const float* __restrict__ ni_buf,
    const float* __restrict__ z_flat, const float* __restrict__ p_buf,
    float* __restrict__ z, float* __restrict__ zw, float* __restrict__ accs, int t)
{
  __shared__ float red[256];
  int gid = blockIdx.x * 256 + threadIdx.x;
  int b = gid >> 10, j = gid & 1023;
  size_t base = (size_t)b * D2_;
  int ci = idxbuf[b];
  float zq_r = codebook[(size_t)ci * D2_ + j];
  float zq_i = codebook[(size_t)ci * D2_ + D_ + j];
  float nr = nr_buf[(size_t)b * D_ + j] + 0.1f * (float)(t + 1);
  float ni = ni_buf[(size_t)b * D_ + j];
  float zr = 0.7f * nr + 0.3f * zq_r;
  float zi = 0.7f * ni + 0.3f * zq_i;
  z[base + j]      = zr;
  z[base + D_ + j] = zi;
  float p = p_buf[b];
  zw[base + j]      += p * zr;
  zw[base + D_ + j] += p * zi;
  float dr = zq_r - z_flat[base + j];
  float di = zq_i - z_flat[base + D_ + j];
  red[threadIdx.x] = dr * dr + di * di; __syncthreads();
  for (int o = 128; o > 0; o >>= 1) {
    if (threadIdx.x < o) red[threadIdx.x] += red[threadIdx.x + o];
    __syncthreads();
  }
  if (threadIdx.x == 0) atomicAdd(&accs[1], red[0]);
}

__global__ void ucrsn_heads(const float* __restrict__ zw, const float* __restrict__ head_W,
                            const float* __restrict__ head_b, float* __restrict__ out)
{
  int gid = blockIdx.x * blockDim.x + threadIdx.x;
  if (gid >= B_ * 7) return;
  int b = gid / 7, h = gid % 7;
  float acc = head_b[h];
  const float* x = zw + (size_t)b * D2_;
  const float* w = head_W + (size_t)h * D2_;
  for (int j = 0; j < D2_; ++j) acc += x[j] * w[j];
  out[gid] = 1.0f / (1.0f + expf(-acc));
}

__global__ void ucrsn_finalize(const float* __restrict__ accs, float* __restrict__ out)
{
  if (blockIdx.x == 0 && threadIdx.x == 0) {
    out[0] = accs[0];                                        // ponder
    out[1] = 1.01f * accs[1] / (float)((size_t)B_ * D2_);    // vq_total (1+COMMIT)
  }
}

// ---------------------------------------------------------------------------
extern "C" void kernel_launch(void* const* d_in, const int* in_sizes, int n_in,
                              void* d_out, int out_size, void* d_ws, size_t ws_size,
                              hipStream_t stream)
{
  (void)in_sizes; (void)n_in; (void)out_size; (void)ws_size;
  const int*   ids       = (const int*)d_in[0];
  const int*   pidx      = (const int*)d_in[1];
  const int*   aidx      = (const int*)d_in[2];
  const int*   olf_idx   = (const int*)d_in[3];
  const int*   gust_idx  = (const int*)d_in[4];
  const float* emb_mag   = (const float*)d_in[5];
  const float* emb_phase = (const float*)d_in[6];
  const float* persp     = (const float*)d_in[7];
  const float* audio     = (const float*)d_in[8];
  const float* olf_emb   = (const float*)d_in[9];
  const float* gust_emb  = (const float*)d_in[10];
  const float* Wr        = (const float*)d_in[11];
  const float* Wi        = (const float*)d_in[12];
  const float* ln_scale  = (const float*)d_in[13];
  const float* ln_shift  = (const float*)d_in[14];
  const float* mod_bias  = (const float*)d_in[15];
  const float* halt_W    = (const float*)d_in[16];
  const float* halt_b    = (const float*)d_in[17];
  const float* codebook  = (const float*)d_in[18];
  const float* adjacency = (const float*)d_in[19];
  const float* dec_W     = (const float*)d_in[20];
  const float* dec_b     = (const float*)d_in[21];
  const float* head_W    = (const float*)d_in[22];
  const float* head_b    = (const float*)d_in[23];

  char* p = (char*)d_ws;
  auto carve = [&](size_t n) { char* r = p; p += (n + 255) & ~(size_t)255; return r; };
  float* z       = (float*)carve((size_t)B_ * D2_ * 4);   // state [zr|zi]
  float* Bs      = (float*)carve((size_t)D2_ * D2_ * 4);  // [Wr|-Wi; Wi|Wr]
  float* lbuf    = (float*)carve((size_t)B_ * D2_ * 4);   // [lr|li]
  float* nr_buf  = (float*)carve((size_t)B_ * D_ * 4);
  float* ni_buf  = (float*)carve((size_t)B_ * D_ * 4);
  float* zflat   = (float*)carve((size_t)B_ * D2_ * 4);
  float* G       = (float*)carve((size_t)B_ * S_ * 4);    // z_flat @ codebook^T
  float* zw      = (float*)carve((size_t)B_ * D2_ * 4);   // ACT-weighted output
  float* cbsq    = (float*)carve((size_t)S_ * 4);
  float* phalt   = (float*)carve((size_t)B_ * 4);
  float* pbuf    = (float*)carve((size_t)B_ * 4);
  float* halting = (float*)carve((size_t)B_ * 4);
  float* remain  = (float*)carve((size_t)B_ * 4);
  int*   idxbuf  = (int*)carve((size_t)B_ * 4);
  float* accs    = (float*)carve(2 * 4);                  // [ponder, vq_sum]

  dim3 blk(256);
  ucrsn_init<<<(B_ * D_ + 255) / 256, blk, 0, stream>>>(
      ids, pidx, aidx, emb_mag, emb_phase, persp, audio, z, zw, halting, remain, accs);
  ucrsn_bstack<<<(D_ * D_ + 255) / 256, blk, 0, stream>>>(Wr, Wi, Bs);
  ucrsn_cbsq<<<S_, blk, 0, stream>>>(codebook, cbsq);

  for (int t = 0; t < TMX; ++t) {
    // complex rotation: [lr|li] = z @ Bs^T   (M=2048, N=2048, K=2048)
    ucrsn_gemm_f16<false><<<dim3(D2_ / BN, B_ / BM), blk, 0, stream>>>(
        z, Bs, lbuf, nullptr, B_, D2_, D2_);
    ucrsn_recurrent<<<B_, blk, 0, stream>>>(
        lbuf, ln_scale, ln_shift, mod_bias, halt_W, halt_b,
        olf_idx, gust_idx, olf_emb, gust_emb, nr_buf, ni_buf, zflat, phalt, t);
    // codebook Gram: G = z_flat @ codebook^T  (M=2048, N=8192, K=2048; cb L2-resident)
    ucrsn_gemm_f16<false><<<dim3(S_ / BN, B_ / BM), blk, 0, stream>>>(
        zflat, codebook, G, nullptr, B_, S_, D2_);
    ucrsn_argmin_act<<<B_, blk, 0, stream>>>(
        G, cbsq, adjacency, idxbuf, phalt, pbuf, halting, remain, accs, t);
    ucrsn_update<<<(B_ * D_) / 256, blk, 0, stream>>>(
        codebook, idxbuf, nr_buf, ni_buf, zflat, pbuf, z, zw, accs, t);
  }

  // decoder: logits = zw @ dec_W^T + dec_b  (M=2048, N=50257, K=2048)
  float* logits = (float*)d_out;
  ucrsn_gemm_f16<true><<<dim3((V_ + BN - 1) / BN, B_ / BM), blk, 0, stream>>>(
      zw, dec_W, logits, dec_b, B_, V_, D2_);
  float* feats = logits + (size_t)B_ * V_;
  ucrsn_heads<<<(B_ * 7 + 255) / 256, blk, 0, stream>>>(zw, head_W, head_b, feats);
  ucrsn_finalize<<<1, 1, 0, stream>>>(accs, feats + (size_t)B_ * 7);
}
